// GCMCLayer_70549132804177
// MI455X (gfx1250) — matrix operand order repaired and checked by reading, not compiled, not run on hardware
//
#include <hip/hip_runtime.h>
#include <hip/hip_bf16.h>

typedef __attribute__((ext_vector_type(16))) _Float16 v16h;
typedef __attribute__((ext_vector_type(8)))  float    v8f;

#define R_   5
#define NU_  100000
#define NM_  100000
#define E_   1000000
#define IN_  64
#define MSG_ 80
#define OUT_ 64
#define BAS_ 4
#define MPR_ 16

// ---------------- zero the aggregation buffers ----------------
__global__ void gcmc_zero(float4* p, long long n4) {
    long long i = (long long)blockIdx.x * blockDim.x + threadIdx.x;
    if (i < n4) p[i] = make_float4(0.f, 0.f, 0.f, 0.f);
}

// ---------------- prep: W = att.basis, swizzled into WMMA-B lane layout ----
// Wb layout:  [r][ks(2)][lane(32)][j(16)] f16 ; lane = hi*16+col, element j = W[r][ks*32+hi*16+j][col]
// Bfc layout: [otile(4)][ks(3)][lane(32)][j(16)] f16 ; value fc_w[otile*16+col][ks*32+hi*16+j] (0-padded K>=80)
__global__ void gcmc_prep(const float* __restrict__ att, const float* __restrict__ basis,
                          const float* __restrict__ fc_w,
                          _Float16* __restrict__ Wb, _Float16* __restrict__ Bfc) {
    int tid = blockIdx.x * blockDim.x + threadIdx.x;
    if (tid < R_ * IN_ * MPR_) {
        int r = tid / (IN_ * MPR_);
        int rem = tid % (IN_ * MPR_);
        int k = rem / MPR_;
        int o = rem % MPR_;
        float acc = 0.f;
        #pragma unroll
        for (int b = 0; b < BAS_; ++b)
            acc += att[r * BAS_ + b] * basis[(b * IN_ + k) * MPR_ + o];
        int ks = k >> 5, kk = k & 31, hi = kk >> 4, j = kk & 15;
        int lane = hi * 16 + o;
        Wb[(((r * 2 + ks) * 32) + lane) * 16 + j] = (_Float16)acc;
    } else {
        int t = tid - R_ * IN_ * MPR_;
        if (t < 4 * 3 * 32 * 16) {
            int rem = t;
            int j    = rem & 15;          rem >>= 4;
            int lane = rem & 31;          rem >>= 5;
            int ks   = rem % 3;
            int otile= rem / 3;
            int o  = otile * 16 + (lane & 15);
            int hi = lane >> 4;
            int k  = ks * 32 + hi * 16 + j;
            float v = (k < MSG_) ? fc_w[o * MSG_ + k] : 0.f;
            Bfc[t] = (_Float16)v;
        }
    }
}

// ---------------- feature GEMM: f[r][n][o] = (feat[n]*cj[n]) @ W[r] --------
// one wave -> one (type, rating, 16-node tile), K=64 in two 32-K WMMA steps
__global__ void gcmc_feat_wmma(const float* __restrict__ ufeat, const float* __restrict__ ifeat,
                               const float* __restrict__ cj_user, const float* __restrict__ cj_movie,
                               const _Float16* __restrict__ Wb,
                               float* __restrict__ fu, float* __restrict__ fi) {
    int wave = (blockIdx.x * blockDim.x + threadIdx.x) >> 5;
    int lane = threadIdx.x & 31;
    const int tilesPer = NU_ / 16;                 // 6250
    const int wavesPerType = R_ * tilesPer;        // 31250
    if (wave >= 2 * wavesPerType) return;          // wave-uniform: EXEC stays all-ones
    int type = wave / wavesPerType;
    int w    = wave % wavesPerType;
    int r    = w / tilesPer;
    int tile = w % tilesPer;

    const float* feat = type ? ifeat : ufeat;
    const float* cj   = type ? cj_movie : cj_user;
    float*       outp = type ? fi : fu;

    int row = tile * 16 + (lane & 15);
    int hi  = lane >> 4;
    float scale = cj[row];

    v8f c = {};
    #pragma unroll
    for (int ks = 0; ks < 2; ++ks) {
        // A operand (16x32 f16): lanes<16 hold K {0..15}, lanes>=16 hold K {16..31}
        const float* fr = feat + row * IN_ + ks * 32;
        v16h a;
        #pragma unroll
        for (int j = 0; j < 8; ++j) a[j]     = (_Float16)(fr[hi * 16 + j] * scale);
        #pragma unroll
        for (int j = 0; j < 8; ++j) a[8 + j] = (_Float16)(fr[8 + hi * 16 + j] * scale);
        // B operand (32x16 f16), pre-swizzled: contiguous 32B per lane
        v16h b = *(const v16h*)(Wb + (((r * 2 + ks) * 32) + lane) * 16);
        c = __builtin_amdgcn_wmma_f32_16x16x32_f16(false, a, false, b, (short)0, c, false, false);
    }
    // C layout: VGPR v -> row (hi*8+v), col (lane&15)
    int col = lane & 15;
    #pragma unroll
    for (int v = 0; v < 8; ++v) {
        int m = tile * 16 + hi * 8 + v;
        outp[((long long)r * NU_ + m) * MPR_ + col] = c[v];
    }
}

// ---------------- edge scatter: agg[dst][r*16+k] += fsrc[r][src][k] --------
// 16 lanes per edge -> 64B coalesced gather + one f32 atomic per lane (L2-resident)
__global__ void gcmc_scatter(const int* __restrict__ edge_src, const int* __restrict__ edge_dst,
                             const float* __restrict__ fsrc, float* __restrict__ agg) {
    long long idx = (long long)blockIdx.x * blockDim.x + threadIdx.x;
    if (idx >= (long long)R_ * E_ * MPR_) return;
    int k = (int)(idx & 15);
    long long e = idx >> 4;            // 0 .. R*E-1 ; flat index into edge arrays
    int r = (int)(e / E_);
    int src = edge_src[e];
    int dst = edge_dst[e];
    float v = fsrc[((long long)r * NU_ + src) * MPR_ + k];
    unsafeAtomicAdd(&agg[(long long)dst * MSG_ + r * MPR_ + k], v);
}

// ---------------- final FC: out = relu(agg*ci) @ fc_w.T + b ---------------
// one wave -> 16 nodes x 64 outputs: A reused across 4 column tiles, K=80 in 3 steps
__global__ void gcmc_fc_wmma(const float* __restrict__ u_agg, const float* __restrict__ m_agg,
                             const float* __restrict__ ci_user, const float* __restrict__ ci_movie,
                             const _Float16* __restrict__ Bfc, const float* __restrict__ fc_b,
                             float* __restrict__ out) {
    int wave = (blockIdx.x * blockDim.x + threadIdx.x) >> 5;
    int lane = threadIdx.x & 31;
    const int tiles = (NU_ + NM_) / 16;            // 12500
    if (wave >= tiles) return;                     // wave-uniform guard
    int nodeBase = wave * 16;

    const float* agg; const float* ci; float* optr; int nb;
    if (nodeBase < NU_) { agg = u_agg; ci = ci_user;  optr = out;                         nb = nodeBase; }
    else                { agg = m_agg; ci = ci_movie; optr = out + (long long)NU_ * OUT_; nb = nodeBase - NU_; }

    int rrow = nb + (lane & 15);
    int hi   = lane >> 4;
    float s  = ci[rrow];

    v8f acc[4] = {{}, {}, {}, {}};
    #pragma unroll
    for (int ks = 0; ks < 3; ++ks) {
        const float* ar = agg + (long long)rrow * MSG_;
        v16h a;
        #pragma unroll
        for (int j = 0; j < 8; ++j) {
            int k = ks * 32 + hi * 16 + j;
            float v = (k < MSG_) ? ar[k] * s : 0.f;
            a[j] = (_Float16)(v > 0.f ? v : 0.f);
        }
        #pragma unroll
        for (int j = 0; j < 8; ++j) {
            int k = ks * 32 + 8 + hi * 16 + j;
            float v = (k < MSG_) ? ar[k] * s : 0.f;
            a[8 + j] = (_Float16)(v > 0.f ? v : 0.f);
        }
        #pragma unroll
        for (int ot = 0; ot < 4; ++ot) {
            v16h b = *(const v16h*)(Bfc + ((ot * 3 + ks) * 32 + lane) * 16);
            acc[ot] = __builtin_amdgcn_wmma_f32_16x16x32_f16(false, a, false, b, (short)0, acc[ot], false, false);
        }
    }
    int col = lane & 15;
    #pragma unroll
    for (int ot = 0; ot < 4; ++ot) {
        float bias = fc_b[ot * 16 + col];
        #pragma unroll
        for (int v = 0; v < 8; ++v) {
            int m = nb + hi * 8 + v;
            optr[(long long)m * OUT_ + ot * 16 + col] = acc[ot][v] + bias;
        }
    }
}

extern "C" void kernel_launch(void* const* d_in, const int* in_sizes, int n_in,
                              void* d_out, int out_size, void* d_ws, size_t ws_size,
                              hipStream_t stream) {
    const float* ufeat    = (const float*)d_in[0];
    const float* ifeat    = (const float*)d_in[1];
    const float* cj_user  = (const float*)d_in[2];
    const float* ci_user  = (const float*)d_in[3];
    const float* cj_movie = (const float*)d_in[4];
    const float* ci_movie = (const float*)d_in[5];
    const int*   edge_user  = (const int*)d_in[6];
    const int*   edge_movie = (const int*)d_in[7];
    const float* att   = (const float*)d_in[8];
    const float* basis = (const float*)d_in[9];
    const float* fc_w  = (const float*)d_in[10];
    const float* fc_b  = (const float*)d_in[11];
    float* out = (float*)d_out;

    char* ws = (char*)d_ws;
    _Float16* Wb  = (_Float16*)(ws);                       //  10,240 B
    _Float16* Bfc = (_Float16*)(ws + 16384);               //  12,288 B
    float* fu    = (float*)(ws + 32768);                   //  32 MB  [R][NU][16]
    float* fi    = (float*)(ws + 32768 +  32000000LL);     //  32 MB  [R][NM][16]
    float* u_agg = (float*)(ws + 32768 +  64000000LL);     //  32 MB  [NU][80]
    float* m_agg = (float*)(ws + 32768 +  96000000LL);     //  32 MB  [NM][80]

    // 1) zero both agg buffers (contiguous 64 MB)
    {
        long long n4 = (long long)(NU_ + NM_) * MSG_ / 4;  // 4,000,000 float4
        int blocks = (int)((n4 + 255) / 256);
        gcmc_zero<<<blocks, 256, 0, stream>>>((float4*)u_agg, n4);
    }
    // 2) basis combination + operand swizzles
    {
        int total = R_ * IN_ * MPR_ + 4 * 3 * 32 * 16;     // 5120 + 6144
        gcmc_prep<<<(total + 255) / 256, 256, 0, stream>>>(att, basis, fc_w, Wb, Bfc);
    }
    // 3) per-rating feature GEMMs with cj folded in (WMMA)
    {
        int waves = 2 * R_ * (NU_ / 16);                   // 62500
        int blocks = (waves * 32 + 255) / 256;
        gcmc_feat_wmma<<<blocks, 256, 0, stream>>>(ufeat, ifeat, cj_user, cj_movie, Wb, fu, fi);
    }
    // 4) edge scatter, both directions (f32 atomics, L2-resident)
    {
        long long thr = (long long)R_ * E_ * MPR_;         // 80,000,000
        int blocks = (int)((thr + 255) / 256);
        gcmc_scatter<<<blocks, 256, 0, stream>>>(edge_user, edge_movie, fu, m_agg);
        gcmc_scatter<<<blocks, 256, 0, stream>>>(edge_movie, edge_user, fi, u_agg);
    }
    // 5) fused ci * relu * FC + bias (WMMA), writes u then m into d_out
    {
        int waves = (NU_ + NM_) / 16;                      // 12500
        int blocks = (waves * 32 + 255) / 256;
        gcmc_fc_wmma<<<blocks, 256, 0, stream>>>(u_agg, m_agg, ci_user, ci_movie, Bfc, fc_b, out);
    }
}
